// CustomLiltSelfAttention_73495480369240
// MI455X (gfx1250) — matrix operand order repaired and checked by reading, not compiled, not run on hardware
//
#include <hip/hip_runtime.h>
#include <stdint.h>

// ---------------------------------------------------------------------------
// CDNA5 / gfx1250 LiLT self-attention (bf16 WMMA + async global->LDS DMA)
// B=4, S=1024, D=1024, NH=16, HD=64, LD=256, LHD=16, MAXP=2048
// ---------------------------------------------------------------------------

typedef __attribute__((ext_vector_type(16))) __bf16    v16bf;
typedef __attribute__((ext_vector_type(8)))  float     v8f;
typedef __attribute__((ext_vector_type(4)))  uint32_t  v4u;

union FragB { v16bf v; uint16_t u[16]; uint32_t d[8]; v4u q[2]; };
union FragC { v8f   v; float    f[8];  };

__device__ __forceinline__ uint16_t f2bf(float x) {
  union { float f; uint32_t i; } c; c.f = x;
  uint32_t r = c.i + 0x7FFFu + ((c.i >> 16) & 1u);  // round-to-nearest-even
  return (uint16_t)(r >> 16);
}
__device__ __forceinline__ uint32_t pk2bf(float lo, float hi) {
  return (uint32_t)f2bf(lo) | ((uint32_t)f2bf(hi) << 16);
}

#define WMMA_BF16(A, B, C) \
  __builtin_amdgcn_wmma_f32_16x16x32_bf16(false, (A), false, (B), (short)0, (C), false, false)

// ds_swizzle xor-butterfly (group-of-32: and=0x1F, or=0, xor at offset[14:10])
template<int OFF>
__device__ __forceinline__ float swz(float v) {
  union { float f; int i; } c; c.f = v;
  c.i = __builtin_amdgcn_ds_swizzle(c.i, OFF);
  return c.f;
}
__device__ __forceinline__ float wave16_max(float v) {
  v = fmaxf(v, swz<0x041F>(v));
  v = fmaxf(v, swz<0x081F>(v));
  v = fmaxf(v, swz<0x101F>(v));
  v = fmaxf(v, swz<0x201F>(v));
  return v;
}
__device__ __forceinline__ float wave16_sum(float v) {
  v += swz<0x041F>(v);
  v += swz<0x081F>(v);
  v += swz<0x101F>(v);
  v += swz<0x201F>(v);
  return v;
}

// -- CDNA5 async global->LDS copies (ASYNCcnt), per CDNA5 ISA --
__device__ __forceinline__ void async_g2l_b128(void* lds, const void* gsrc) {
  asm volatile("global_load_async_to_lds_b128 %0, %1, off"
               :: "v"((uint32_t)(uintptr_t)lds), "v"((uint64_t)(uintptr_t)gsrc)
               : "memory");
}
__device__ __forceinline__ void async_g2l_b32(void* lds, const void* gsrc) {
  asm volatile("global_load_async_to_lds_b32 %0, %1, off"
               :: "v"((uint32_t)(uintptr_t)lds), "v"((uint64_t)(uintptr_t)gsrc)
               : "memory");
}
__device__ __forceinline__ void wait_async0() {
  asm volatile("s_wait_asynccnt 0x0" ::: "memory");
}

// ---------------------------------------------------------------------------
// Projection GEMM: out[b,h,s,d] = bf16( X[m,:] @ W[:,n] + bias[n] )
// Tile 64x64, K-step 32, 8 waves (4M x 2N), each wave 16x32.
// ---------------------------------------------------------------------------
template<int KD, int NN, int SS, int HDP>
__global__ __launch_bounds__(256)
void lilt_proj_gemm(const float* __restrict__ X, const float* __restrict__ W,
                    const float* __restrict__ bias, uint16_t* __restrict__ out)
{
  __shared__ uint16_t As[64][32];   // row-major bf16
  __shared__ uint32_t Bp[64][16];   // [n][kp]: pack(W[2kp][n], W[2kp+1][n])

  const int tid  = threadIdx.x;
  const int lane = tid & 31;
  const int wave = tid >> 5;
  const int half = lane >> 4, ln = lane & 15;
  const int wm = wave & 3, wn = wave >> 2;
  const int mBase = blockIdx.x * 64;
  const int nBase = blockIdx.y * 64;

  FragC acc[2];
  for (int t = 0; t < 2; ++t)
    for (int r = 0; r < 8; ++r) acc[t].f[r] = 0.0f;

  for (int k0 = 0; k0 < KD; k0 += 32) {
    __syncthreads();
    {   // A tile 64x32: one v4u (4 packed dwords) per thread
      const int row = tid >> 2, c0 = tid & 3;            // v4u col
      const float* src = X + (size_t)(mBase + row) * KD + k0 + c0 * 8;
      v4u* arow = (v4u*)&As[row][0];
      v4u pk = { pk2bf(src[0], src[1]), pk2bf(src[2], src[3]),
                 pk2bf(src[4], src[5]), pk2bf(src[6], src[7]) };
      arow[c0] = pk;
    }
    {   // B tile 32x64 -> packed [n][kp], one v4u per thread
      const int n = tid & 63, kp0 = (tid >> 6) * 4;
      v4u pk;
      #pragma unroll
      for (int u = 0; u < 4; ++u) {
        const int kp = kp0 + u;
        pk[u] = pk2bf(W[(size_t)(k0 + 2 * kp) * NN + nBase + n],
                      W[(size_t)(k0 + 2 * kp + 1) * NN + nBase + n]);
      }
      *(v4u*)&Bp[n][kp0] = pk;
    }
    __syncthreads();

    FragB a;
    const v4u* arow = (const v4u*)&As[wm * 16 + ln][0];
    a.q[0] = arow[2 * half];
    a.q[1] = arow[2 * half + 1];
    #pragma unroll
    for (int t = 0; t < 2; ++t) {
      FragB bf;
      const v4u* brow = (const v4u*)&Bp[wn * 32 + t * 16 + ln][0];
      bf.q[0] = brow[2 * half];
      bf.q[1] = brow[2 * half + 1];
      acc[t].v = WMMA_BF16(a.v, bf.v, acc[t].v);
    }
  }

  constexpr int NH = NN / HDP;
  #pragma unroll
  for (int t = 0; t < 2; ++t) {
    #pragma unroll
    for (int r = 0; r < 8; ++r) {
      const int m = mBase + wm * 16 + r + 8 * half;
      const int n = nBase + wn * 32 + t * 16 + ln;
      const int bb = m / SS, ss = m % SS;        // constexpr -> shifts
      const int hh = n / HDP, dd = n % HDP;
      out[(((size_t)bb * NH + hh) * SS + ss) * HDP + dd] = f2bf(acc[t].f[r] + bias[n]);
    }
  }
}

// ---------------------------------------------------------------------------
// Fused attention. Grid: (S/64, B*NH). Block: 128 threads = 4 waves.
// Per key tile: 2 WMMA (QK) + 1 (LQ.LK) + 4 (pos-bias GEMM) + 4 (PV) + 1 (P.LV)
// Padded-K tiles keep zeros in LDS (upper 8 dwords of each row), so every
// fragment build is exactly two unconditional 16B LDS loads.
// ---------------------------------------------------------------------------
__global__ __launch_bounds__(128)
void lilt_attn(const uint16_t* __restrict__ Q, const uint16_t* __restrict__ K,
               const uint16_t* __restrict__ V, const uint16_t* __restrict__ LQ,
               const uint16_t* __restrict__ LK, const uint16_t* __restrict__ LV,
               const float* __restrict__ mask, const float* __restrict__ dist,
               float* __restrict__ ctx, float* __restrict__ lctx)
{
  constexpr int S = 1024, HD = 64, LHD = 16, D = 1024, LD = 256;

  __shared__ uint16_t Kt[16][64];     // natural rows of K tile (async DMA)
  __shared__ uint16_t LKt[16][32];    // rows: real u16 0..15, zero 16..31
  __shared__ uint32_t Vp[64][16];     // [d][rp] packed pairs; dwords 8..15 zero
  __shared__ uint32_t LVp[16][16];    // [d][rp]; dwords 8..15 zero
  __shared__ uint16_t Eb[4][32][64];  // per-wave dist_emb band rows (bf16)
  __shared__ float    Gb[4][16][32];  // per-wave G = Q @ Eband^T
  __shared__ uint16_t Pb[4][16][32];  // per-wave probs; dwords 8..15 zero

  const int tid = threadIdx.x, lane = tid & 31, w = tid >> 5;
  const int half = lane >> 4, ln = lane & 15;
  const int bh = blockIdx.y, b = bh >> 4, h = bh & 15;
  const int l0 = blockIdx.x * 64 + w * 16;

  const uint16_t* Qb  = Q  + (size_t)bh * S * HD;
  const uint16_t* Kb  = K  + (size_t)bh * S * HD;
  const uint16_t* Vb  = V  + (size_t)bh * S * HD;
  const uint16_t* LQb = LQ + (size_t)bh * S * LHD;
  const uint16_t* LKb = LK + (size_t)bh * S * LHD;
  const uint16_t* LVb = LV + (size_t)bh * S * LHD;

  // ---- zero the padded-K regions once (never rewritten) ----
  {
    uint32_t* vz = &Vp[0][0];
    #pragma unroll
    for (int u = 0; u < 4; ++u) {                // 64 rows x 8 dwords
      const int e = tid + 128 * u;
      vz[(e >> 3) * 16 + 8 + (e & 7)] = 0u;
    }
    LVp[tid >> 3][8 + (tid & 7)] = 0u;           // 16 rows x 8 dwords
    ((uint32_t*)LKt)[(tid >> 3) * 16 + 8 + (tid & 7)] = 0u;
    uint32_t* pz = (uint32_t*)Pb;                // 4*16 rows x 8 dwords
    #pragma unroll
    for (int u = 0; u < 4; ++u) {
      const int e = tid + 128 * u;
      pz[(e >> 3) * 16 + 8 + (e & 7)] = 0u;
    }
  }

  // ---- Q / LQ A-fragments (held in registers for the whole pass) ----
  FragB aQ0, aQ1, aLQ;
  {
    const v4u* qv = (const v4u*)(Qb  + (size_t)(l0 + ln) * HD);   // 8 v4u/row
    const v4u* lv = (const v4u*)(LQb + (size_t)(l0 + ln) * LHD);  // 2 v4u/row
    aQ0.q[0] = qv[2 * half];     aQ0.q[1] = qv[2 * half + 1];
    aQ1.q[0] = qv[4 + 2 * half]; aQ1.q[1] = qv[5 + 2 * half];
    aLQ.q[0] = lv[half];     aLQ.q[1] = (v4u)(0u);   // K padded 16->32
  }

  FragC cacc[4], lacc;
  #pragma unroll
  for (int t = 0; t < 4; ++t)
    for (int r = 0; r < 8; ++r) cacc[t].f[r] = 0.0f;
  for (int r = 0; r < 8; ++r) lacc.f[r] = 0.0f;
  float mst[8], lst[8];
  for (int r = 0; r < 8; ++r) { mst[r] = -1e30f; lst[r] = 0.0f; }

  for (int j = 0; j < 64; ++j) {
    const int r0 = j * 16;
    __syncthreads();

    // ---- async DMA: K tile (16B/thread) and LK tile (4B/thread) ----
    {
      const int rr = tid >> 3, db = (tid & 7) * 8;
      async_g2l_b128(&Kt[rr][db], Kb + (size_t)(r0 + rr) * HD + db);
      const int db2 = (tid & 7) * 2;
      async_g2l_b32(&LKt[rr][db2], LKb + (size_t)(r0 + rr) * LHD + db2);
    }
    // ---- overlapped VALU pack loads: V, LV (pairs over r), E band ----
    {
      const int d = tid >> 1, rp0 = (tid & 1) * 4;
      v4u pk;
      #pragma unroll
      for (int u = 0; u < 4; ++u) {
        const int rp = rp0 + u;
        pk[u] = (uint32_t)Vb[(size_t)(r0 + 2 * rp) * HD + d] |
                ((uint32_t)Vb[(size_t)(r0 + 2 * rp + 1) * HD + d] << 16);
      }
      *(v4u*)&Vp[d][rp0] = pk;
      const int d2 = tid & 15, rp2 = tid >> 4;
      LVp[d2][rp2] = (uint32_t)LVb[(size_t)(r0 + 2 * rp2) * LHD + d2] |
                     ((uint32_t)LVb[(size_t)(r0 + 2 * rp2 + 1) * LHD + d2] << 16);
      // dist_emb band rows (2047+l0-r0)-15 .. +16  (always in [1024,3071])
      const int er = 2032 + l0 - r0 + lane;
      const float* es = dist + (size_t)er * HD;
      v4u* erow = (v4u*)&Eb[w][lane][0];
      #pragma unroll
      for (int u = 0; u < 8; ++u) {
        v4u pkx = { pk2bf(es[8 * u + 0], es[8 * u + 1]), pk2bf(es[8 * u + 2], es[8 * u + 3]),
                    pk2bf(es[8 * u + 4], es[8 * u + 5]), pk2bf(es[8 * u + 6], es[8 * u + 7]) };
        erow[u] = pkx;
      }
    }
    wait_async0();
    __syncthreads();

    // ---- B fragments: two b128 LDS loads each, no selects ----
    FragB bK0, bK1, bLK, bV0, bV1, bV2, bV3, bLV, bE00, bE01, bE10, bE11;
    {
      const v4u* kr  = (const v4u*)&Kt[ln][0];        // 8 v4u/row
      const v4u* lkr = (const v4u*)&LKt[ln][0];       // 4 v4u/row
      const v4u* v0  = (const v4u*)&Vp[ln][0];        // 4 v4u/row
      const v4u* v1  = (const v4u*)&Vp[16 + ln][0];
      const v4u* v2  = (const v4u*)&Vp[32 + ln][0];
      const v4u* v3  = (const v4u*)&Vp[48 + ln][0];
      const v4u* lvr = (const v4u*)&LVp[ln][0];
      const v4u* e0  = (const v4u*)&Eb[w][ln][0];     // 8 v4u/row
      const v4u* e1  = (const v4u*)&Eb[w][16 + ln][0];
      const int hb = 2 * half;
      bK0.q[0] = kr[hb];      bK0.q[1] = kr[hb + 1];
      bK1.q[0] = kr[4 + hb];  bK1.q[1] = kr[5 + hb];
      bLK.q[0] = lkr[half];   bLK.q[1] = lkr[2 + half]; // q[1] zeros
      bV0.q[0] = v0[hb];      bV0.q[1] = v0[hb + 1];
      bV1.q[0] = v1[hb];      bV1.q[1] = v1[hb + 1];
      bV2.q[0] = v2[hb];      bV2.q[1] = v2[hb + 1];
      bV3.q[0] = v3[hb];      bV3.q[1] = v3[hb + 1];
      bLV.q[0] = lvr[hb];     bLV.q[1] = lvr[hb + 1];
      bE00.q[0] = e0[hb];     bE00.q[1] = e0[hb + 1];
      bE01.q[0] = e0[4 + hb]; bE01.q[1] = e0[5 + hb];
      bE10.q[0] = e1[hb];     bE10.q[1] = e1[hb + 1];
      bE11.q[0] = e1[4 + hb]; bE11.q[1] = e1[5 + hb];
    }

    // ---- scores ----
    FragC sc, sl, g0, g1;
    for (int r = 0; r < 8; ++r) { sc.f[r] = 0.f; sl.f[r] = 0.f; g0.f[r] = 0.f; g1.f[r] = 0.f; }
    sc.v = WMMA_BF16(aQ0.v, bK0.v, sc.v);
    sc.v = WMMA_BF16(aQ1.v, bK1.v, sc.v);
    sl.v = WMMA_BF16(aLQ.v, bLK.v, sl.v);
    g0.v = WMMA_BF16(aQ0.v, bE00.v, g0.v);
    g0.v = WMMA_BF16(aQ1.v, bE01.v, g0.v);
    g1.v = WMMA_BF16(aQ0.v, bE10.v, g1.v);
    g1.v = WMMA_BF16(aQ1.v, bE11.v, g1.v);
    #pragma unroll
    for (int r = 0; r < 8; ++r) {
      const int M = r + 8 * half;
      Gb[w][M][ln]      = g0.f[r];
      Gb[w][M][16 + ln] = g1.f[r];
    }
    __syncthreads();

    // ---- combine + online softmax (ds_swizzle butterflies) ----
    const float mv = mask[(size_t)b * S + r0 + ln];
    #pragma unroll
    for (int r = 0; r < 8; ++r) {
      const int M = r + 8 * half;
      const float pbias = Gb[w][M][15 + M - ln];       // G[i][15+i-j]
      const float val = (sc.f[r] + pbias) * 0.125f + sl.f[r] * 0.25f + mv;
      const float rm   = wave16_max(val);
      const float mnew  = fmaxf(mst[r], rm);
      const float alpha = __expf(mst[r] - mnew);
      mst[r] = mnew;
      const float p = __expf(val - mnew);
      lst[r] = lst[r] * alpha + wave16_sum(p);
      #pragma unroll
      for (int t = 0; t < 4; ++t) cacc[t].f[r] *= alpha;
      lacc.f[r] *= alpha;
      Pb[w][M][ln] = f2bf(p);
    }
    __syncthreads();

    // ---- P fragment (A-layout, K padded 16->32): two b128, zeros from LDS ----
    FragB aP;
    {
      const v4u* pr = (const v4u*)&Pb[w][ln][0];       // 4 v4u/row
      aP.q[0] = pr[2 * half];
      aP.q[1] = pr[2 * half + 1];                      // zero region for half==1
    }
    cacc[0].v = WMMA_BF16(aP.v, bV0.v, cacc[0].v);
    cacc[1].v = WMMA_BF16(aP.v, bV1.v, cacc[1].v);
    cacc[2].v = WMMA_BF16(aP.v, bV2.v, cacc[2].v);
    cacc[3].v = WMMA_BF16(aP.v, bV3.v, cacc[3].v);
    lacc.v    = WMMA_BF16(aP.v, bLV.v, lacc.v);
  }

  // ---- epilogue: normalize and scatter to [B,S,D] / [B,S,LD] fp32 ----
  #pragma unroll
  for (int r = 0; r < 8; ++r) {
    const int M = r + 8 * half;
    const int srow = l0 + M;
    const float inv = 1.0f / lst[r];
    const size_t rb = (size_t)b * S + srow;
    #pragma unroll
    for (int t = 0; t < 4; ++t)
      ctx[rb * D + h * HD + t * 16 + ln] = cacc[t].f[r] * inv;
    lctx[rb * LD + h * LHD + ln] = lacc.f[r] * inv;
  }
}

// ---------------------------------------------------------------------------
extern "C" void kernel_launch(void* const* d_in, const int* in_sizes, int n_in,
                              void* d_out, int out_size, void* d_ws, size_t ws_size,
                              hipStream_t stream)
{
  (void)in_sizes; (void)n_in; (void)out_size; (void)ws_size;

  const float* hidden = (const float*)d_in[0];
  const float* layout = (const float*)d_in[1];
  const float* mask   = (const float*)d_in[2];
  const float* wq  = (const float*)d_in[3];  const float* bq  = (const float*)d_in[4];
  const float* wk  = (const float*)d_in[5];  const float* bk  = (const float*)d_in[6];
  const float* wv  = (const float*)d_in[7];  const float* bv  = (const float*)d_in[8];
  const float* lwq = (const float*)d_in[9];  const float* lbq = (const float*)d_in[10];
  const float* lwk = (const float*)d_in[11]; const float* lbk = (const float*)d_in[12];
  const float* lwv = (const float*)d_in[13]; const float* lbv = (const float*)d_in[14];
  const float* dist = (const float*)d_in[15];

  // workspace: Q,K,V (bf16 [B,NH,S,64]) + LQ,LK,LV (bf16 [B,NH,S,16]) = 31.5 MB
  const size_t QSZ = (size_t)4 * 1024 * 1024;
  const size_t LSZ = (size_t)4 * 1024 * 256;
  uint16_t* ws  = (uint16_t*)d_ws;
  uint16_t* Qd  = ws;
  uint16_t* Kd  = Qd + QSZ;
  uint16_t* Vd  = Kd + QSZ;
  uint16_t* LQd = Vd + QSZ;
  uint16_t* LKd = LQd + LSZ;
  uint16_t* LVd = LKd + LSZ;

  dim3 blk(256);
  dim3 gT(4096 / 64, 1024 / 64);
  lilt_proj_gemm<1024, 1024, 1024, 64><<<gT, blk, 0, stream>>>(hidden, wq, bq, Qd);
  lilt_proj_gemm<1024, 1024, 1024, 64><<<gT, blk, 0, stream>>>(hidden, wk, bk, Kd);
  lilt_proj_gemm<1024, 1024, 1024, 64><<<gT, blk, 0, stream>>>(hidden, wv, bv, Vd);
  dim3 gL(4096 / 64, 256 / 64);
  lilt_proj_gemm<256, 256, 1024, 16><<<gL, blk, 0, stream>>>(layout, lwq, lbq, LQd);
  lilt_proj_gemm<256, 256, 1024, 16><<<gL, blk, 0, stream>>>(layout, lwk, lbk, LKd);
  lilt_proj_gemm<256, 256, 1024, 16><<<gL, blk, 0, stream>>>(layout, lwv, lbv, LVd);

  float* ctx  = (float*)d_out;
  float* lctx = ctx + (size_t)4 * 1024 * 1024;
  dim3 gA(1024 / 64, 4 * 16);
  lilt_attn<<<gA, dim3(128), 0, stream>>>(Qd, Kd, Vd, LQd, LKd, LVd, mask, dist, ctx, lctx);
}
